// UniPhyBlock_22428319219993
// MI455X (gfx1250) — matrix operand order repaired and checked by reading, not compiled, not run on hardware
//
#include <hip/hip_runtime.h>

// ---------------------------------------------------------------------------
// UniPhyBlock on gfx1250 (CDNA5): all GEMM-heavy stages on v_wmma_f32_16x16x32_bf16.
// Fragments assembled with 16-byte vector loads (2 x b128 per fragment):
//   A tiles row-major, B tiles staged TRANSPOSED so lane data is contiguous.
// Conv A tiles staged Global->LDS with global_load_async_to_lds_b128 (ASYNCcnt).
// ---------------------------------------------------------------------------

typedef __attribute__((ext_vector_type(16))) __bf16 v16bf;
typedef __attribute__((ext_vector_type(8)))  float  v8f;
typedef unsigned short u16;
typedef unsigned int   u32;
typedef __attribute__((ext_vector_type(8))) u16 u16x8;

#define B_   2
#define T_   16
#define D_   64
#define HH_  64
#define WW_  64
#define HW_  4096          // H*W
#define NPIX 131072        // B*T*H*W  (== token count)
#define CH_  128           // 2*D
#define HID_ 128
#define OUTHALF 8388608    // B*T*D*H*W

__device__ __forceinline__ u16 f32_to_bf16(float f) {
  return __builtin_bit_cast(u16, (__bf16)f);   // native v_cvt, RNE
}
__device__ __forceinline__ float bf16_to_f32(u16 h) {
  u32 x = ((u32)h) << 16;
  return __builtin_bit_cast(float, x);
}
__device__ __forceinline__ u32 pack2(u16 lo, u16 hi) {
  return (u32)lo | ((u32)hi << 16);
}

// CDNA5 async Global->LDS copy (16B per lane), tracked by ASYNCcnt.
__device__ __forceinline__ void async_load_b128(u32 lds_addr, const void* gaddr) {
  asm volatile("global_load_async_to_lds_b128 %0, %1, off"
               :: "v"(lds_addr), "v"(gaddr) : "memory");
}
__device__ __forceinline__ void wait_async0() {
  asm volatile("s_wait_asynccnt 0x0" ::: "memory");
}

union FragV { v16bf v; u16x8 h[2]; };

__device__ __forceinline__ v8f zero8() {
  v8f z = {0.f, 0.f, 0.f, 0.f, 0.f, 0.f, 0.f, 0.f};
  return z;
}

// A fragment (16x32 bf16, ISA 7.12.2): lane L, m=L&15, halves j=0..7 hold
// K=hi8+0..7, halves 8..15 hold K=hi8+16..23 (hi8 = 8*(L>=16)).
__device__ __forceinline__ v16bf fragA_rm(const u16* base, int stride, int lane) {
  int m = lane & 15, hi8 = (lane >> 4) << 3;
  const u16* p = base + m * stride + hi8;
  FragV f;
  f.h[0] = *(const u16x8*)(p);
  f.h[1] = *(const u16x8*)(p + 16);
  return f.v;
}

// B fragment (32x16 bf16): lane L, n=L&15, halves j=0..7 hold K=khi+j, j=8..15 hold K=khi+16+(j-8).
// Tile staged TRANSPOSED: element (k,n) at baseT[n*stride+k].
__device__ __forceinline__ v16bf fragB_tr(const u16* baseT, int stride, int lane) {
  int n = lane & 15, khi = (lane >> 4) << 3;
  const u16* p = baseT + n * stride + khi;
  FragV f;
  f.h[0] = *(const u16x8*)(p);
  f.h[1] = *(const u16x8*)(p + 16);
  return f.v;
}

__device__ __forceinline__ v8f wmma_bf16(v16bf a, v16bf b, v8f c) {
  return __builtin_amdgcn_wmma_f32_16x16x32_bf16(false, a, false, b, (short)0, c,
                                                 false, false);
}

__device__ __forceinline__ float softplus_f(float x) {
  return (x > 0.f) ? (x + log1pf(expf(-x))) : log1pf(expf(x));
}

// ---------------------------------------------------------------------------
// K1: spatial complex LayerNorm -> bf16 conv input  xc[pix][c], c-contiguous
// ---------------------------------------------------------------------------
__global__ void k_ln_spatial(const float* __restrict__ xr, const float* __restrict__ xi,
                             const float* __restrict__ w, const float* __restrict__ bia,
                             u16* __restrict__ xc) {
  int pix = blockIdx.x * blockDim.x + threadIdx.x;     // 0..NPIX-1  (bt, h, w)
  int bt = pix >> 12;
  int hw = pix & 4095;
  size_t base = (size_t)bt * (D_ * HW_) + hw;
  float sum = 0.f, ss = 0.f;
  for (int d = 0; d < D_; ++d) {
    float vr = xr[base + (size_t)d * HW_];
    float vi = xi[base + (size_t)d * HW_];
    sum += vr + vi;
    ss  += vr * vr + vi * vi;
  }
  float mean = sum * (1.f / CH_);
  float var  = ss * (1.f / CH_) - mean * mean;
  float inv  = rsqrtf(var + 1e-5f);
  u32* o32 = (u32*)(xc + (size_t)pix * CH_);
  for (int d = 0; d < D_; d += 2) {
    float vr0 = xr[base + (size_t)d * HW_];
    float vr1 = xr[base + (size_t)(d + 1) * HW_];
    float vi0 = xi[base + (size_t)d * HW_];
    float vi1 = xi[base + (size_t)(d + 1) * HW_];
    u16 r0 = f32_to_bf16((vr0 - mean) * inv * w[d]     + bia[d]);
    u16 r1 = f32_to_bf16((vr1 - mean) * inv * w[d + 1] + bia[d + 1]);
    u16 i0 = f32_to_bf16((vi0 - mean) * inv * w[d + D_]     + bia[d + D_]);
    u16 i1 = f32_to_bf16((vi1 - mean) * inv * w[d + D_ + 1] + bia[d + D_ + 1]);
    o32[d >> 1]        = pack2(r0, r1);
    o32[32 + (d >> 1)] = pack2(i0, i1);
  }
}

// ---------------------------------------------------------------------------
// K2: 3x3 SAME conv 128->128 as implicit GEMM (WMMA) + bias + residual.
//     Per tap (kh,kw): stage 64x128 A row (async Global->LDS) and full
//     128x128 transposed B once; then 4 K-chunks of WMMA with no barriers.
// ---------------------------------------------------------------------------
__global__ void __launch_bounds__(128) k_conv(
    const u16* __restrict__ xc, const float* __restrict__ cw,
    const float* __restrict__ cb, const float* __restrict__ xr,
    const float* __restrict__ xi, float* __restrict__ x1r,
    float* __restrict__ x1i) {
  __shared__ alignas(16) u16 la[64 * 128];      // 64 pixels x 128 in-ch (one tap)
  __shared__ alignas(16) u16 lbt[128 * 128];    // transposed: [co][ci] (one tap)
  int tid = threadIdx.x;
  int wid = tid >> 5, lane = tid & 31;
  int bt = blockIdx.x >> 6, h = blockIdx.x & 63;

  v8f acc[8];
#pragma unroll
  for (int i = 0; i < 8; ++i) acc[i] = zero8();

  for (int kh = 0; kh < 3; ++kh) {
    int r = h + kh - 1;
    bool rok = (r >= 0) && (r < HH_);
    for (int kw = 0; kw < 3; ++kw) {
      // stage A: pixel p = tid/2, channel half c0; 64 u16 per thread
      {
        int p  = tid >> 1;
        int c0 = (tid & 1) * 64;
        int wc = p + kw - 1;
        u16* dst = la + p * 128 + c0;
        if (rok && (wc >= 0) && (wc < WW_)) {
          const u16* g = xc + ((size_t)(bt * HW_ + r * WW_ + wc) * CH_ + c0);
          u32 ldsa = (u32)(uintptr_t)dst;
#pragma unroll
          for (int q = 0; q < 8; ++q)
            async_load_b128(ldsa + q * 16, g + q * 8);
        } else {
          u16x8 z = {0, 0, 0, 0, 0, 0, 0, 0};
#pragma unroll
          for (int q = 0; q < 8; ++q) ((u16x8*)dst)[q] = z;
        }
      }
      // stage B transposed (thread = out-channel n, coalesced reads per k)
      {
        const float* wsrc = cw + (size_t)((kh * 3 + kw) * CH_) * CH_ + tid;
#pragma unroll 8
        for (int k = 0; k < 128; ++k)
          lbt[tid * 128 + k] = f32_to_bf16(wsrc[(size_t)k * CH_]);
      }
      wait_async0();
      __syncthreads();
#pragma unroll
      for (int kc = 0; kc < 4; ++kc) {
        v16bf a = fragA_rm(la + wid * 16 * 128 + kc * 32, 128, lane);
#pragma unroll
        for (int nb = 0; nb < 8; ++nb) {
          v16bf bm = fragB_tr(lbt + nb * 16 * 128 + kc * 32, 128, lane);
          acc[nb] = wmma_bf16(a, bm, acc[nb]);
        }
      }
      __syncthreads();
    }
  }
  // epilogue: + conv bias + residual, split re/im, layout (B,T,D,H,W)
  int ncol = lane & 15, mhi = (lane >> 4) * 8;
#pragma unroll
  for (int nb = 0; nb < 8; ++nb) {
    int co = nb * 16 + ncol;
    int d  = co & 63;
    float bias = cb[co];
#pragma unroll
    for (int v = 0; v < 8; ++v) {
      int wpix = wid * 16 + mhi + v;
      size_t addr = ((size_t)(bt * D_ + d)) * HW_ + h * WW_ + wpix;
      float val = acc[nb][v] + bias;
      if (co < D_) x1r[addr] = val + xr[addr];
      else         x1i[addr] = val + xi[addr];
    }
  }
}

// ---------------------------------------------------------------------------
// K3: temporal complex LayerNorm -> bf16 xt[(nseq*T+t)*D + d]
// ---------------------------------------------------------------------------
__global__ void k_ln_temporal(const float* __restrict__ x1r, const float* __restrict__ x1i,
                              const float* __restrict__ w, const float* __restrict__ bia,
                              u16* __restrict__ xtr, u16* __restrict__ xti) {
  int pix = blockIdx.x * blockDim.x + threadIdx.x;
  int bt = pix >> 12, hw = pix & 4095;
  int b = bt >> 4, t = bt & 15;
  int nseq = b * HW_ + hw;
  size_t tok = (size_t)nseq * T_ + t;
  size_t base = (size_t)bt * (D_ * HW_) + hw;
  float sum = 0.f, ss = 0.f;
  for (int d = 0; d < D_; ++d) {
    float vr = x1r[base + (size_t)d * HW_];
    float vi = x1i[base + (size_t)d * HW_];
    sum += vr + vi;
    ss  += vr * vr + vi * vi;
  }
  float mean = sum * (1.f / CH_);
  float var  = ss * (1.f / CH_) - mean * mean;
  float inv  = rsqrtf(var + 1e-5f);
  u32* orr = (u32*)(xtr + tok * D_);
  u32* oii = (u32*)(xti + tok * D_);
  for (int d = 0; d < D_; d += 2) {
    float vr0 = x1r[base + (size_t)d * HW_];
    float vr1 = x1r[base + (size_t)(d + 1) * HW_];
    float vi0 = x1i[base + (size_t)d * HW_];
    float vi1 = x1i[base + (size_t)(d + 1) * HW_];
    u16 r0 = f32_to_bf16((vr0 - mean) * inv * w[d]     + bia[d]);
    u16 r1 = f32_to_bf16((vr1 - mean) * inv * w[d + 1] + bia[d + 1]);
    u16 i0 = f32_to_bf16((vi0 - mean) * inv * w[d + D_]     + bia[d + D_]);
    u16 i1 = f32_to_bf16((vi1 - mean) * inv * w[d + D_ + 1] + bia[d + D_ + 1]);
    orr[d >> 1] = pack2(r0, r1);
    oii[d >> 1] = pack2(i0, i1);
  }
}

// ---------------------------------------------------------------------------
// K4: complex encode u = (xt @ E + bias) * forcing + noise   (WMMA x4 per tile)
//     E staged transposed (+ pre-negated imag: bf16 WMMA has no A/B NEG).
// ---------------------------------------------------------------------------
__global__ void __launch_bounds__(128) k_encode(
    const u16* __restrict__ xtr, const u16* __restrict__ xti,
    const float* __restrict__ encr, const float* __restrict__ enci,
    const float* __restrict__ dt, const float* __restrict__ lamr,
    const float* __restrict__ lami, const float* __restrict__ sbr,
    const float* __restrict__ sbi, const float* __restrict__ nzr,
    const float* __restrict__ nzi, float* __restrict__ ur,
    float* __restrict__ ui) {
  __shared__ alignas(16) u16 lert[4096], leit[4096], lenit[4096];  // [n][k]
  int tid = threadIdx.x, wid = tid >> 5, lane = tid & 31;
  for (int i = tid; i < 4096; i += 128) {
    int n = i >> 6, k = i & 63;
    float er = encr[k * D_ + n], ei = enci[k * D_ + n];
    lert[i]  = f32_to_bf16(er);
    leit[i]  = f32_to_bf16(ei);
    lenit[i] = f32_to_bf16(-ei);
  }
  __syncthreads();
  int rowbase = (blockIdx.x * 4 + wid) * 16;

  v8f cr[4], ci[4];
#pragma unroll
  for (int i = 0; i < 4; ++i) { cr[i] = zero8(); ci[i] = zero8(); }

  for (int kc = 0; kc < 2; ++kc) {
    int k0 = kc * 32;
    v16bf ar = fragA_rm(xtr + (size_t)rowbase * D_ + k0, D_, lane);
    v16bf ai = fragA_rm(xti + (size_t)rowbase * D_ + k0, D_, lane);
#pragma unroll
    for (int nb = 0; nb < 4; ++nb) {
      v16bf br  = fragB_tr(lert  + nb * 16 * D_ + k0, D_, lane);
      v16bf bi  = fragB_tr(leit  + nb * 16 * D_ + k0, D_, lane);
      v16bf bni = fragB_tr(lenit + nb * 16 * D_ + k0, D_, lane);
      cr[nb] = wmma_bf16(ar, br,  cr[nb]);
      cr[nb] = wmma_bf16(ai, bni, cr[nb]);
      ci[nb] = wmma_bf16(ar, bi,  ci[nb]);
      ci[nb] = wmma_bf16(ai, br,  ci[nb]);
    }
  }
  // epilogue: ZOH forcing + noise
  int ncol = lane & 15, mhi = (lane >> 4) * 8;
#pragma unroll
  for (int nb = 0; nb < 4; ++nb) {
    int e = nb * 16 + ncol;
    float lr = -softplus_f(lamr[e]);
    float li = lami[e];
    float den = lr * lr + li * li;
    float sbre = sbr[e], sbim = sbi[e];
#pragma unroll
    for (int v = 0; v < 8; ++v) {
      int token = rowbase + mhi + v;
      int nseq = token >> 4, t = token & 15;
      int bidx = nseq >> 12;
      float dtv = dt[bidx * T_ + t];
      float ea  = expf(lr * dtv);
      float ang = li * dtv;
      float arr = ea * cosf(ang);
      float aii = ea * sinf(ang);
      float numr = arr - 1.f, numi = aii;
      float fr = (numr * lr + numi * li) / den;
      float fi = (numi * lr - numr * li) / den;
      float u0r = cr[nb][v] + sbre;
      float u0i = ci[nb][v] + sbim;
      float uur = u0r * fr - u0i * fi;
      float uui = u0r * fi + u0i * fr;
      float sq = 0.01f * sqrtf(dtv);
      size_t idx = (size_t)token * D_ + e;
      uur += sq * nzr[idx];
      uui += sq * nzi[idx];
      ur[idx] = uur;
      ui[idx] = uui;
    }
  }
}

// ---------------------------------------------------------------------------
// K5: first-order linear recurrence over T=16 (per mode); emit bf16 h
// ---------------------------------------------------------------------------
__global__ void k_scan(const float* __restrict__ ur, const float* __restrict__ ui,
                       const float* __restrict__ dt, const float* __restrict__ lamr,
                       const float* __restrict__ lami,
                       u16* __restrict__ hrb, u16* __restrict__ hib) {
  int idx = blockIdx.x * blockDim.x + threadIdx.x;   // NSEQ*D
  int e = idx & 63, nseq = idx >> 6;
  int bidx = nseq >> 12;
  float lr = -softplus_f(lamr[e]);
  float li = lami[e];
  float hr = 0.f, hi = 0.f;
  for (int t = 0; t < T_; ++t) {
    float dtv = dt[bidx * T_ + t];
    float ea  = expf(lr * dtv);
    float arr = ea * cosf(li * dtv);
    float aii = ea * sinf(li * dtv);
    size_t p = ((size_t)nseq * T_ + t) * D_ + e;
    float uur = ur[p], uui = ui[p];
    float nhr = arr * hr - aii * hi + uur;
    float nhi = arr * hi + aii * hr + uui;
    hr = nhr; hi = nhi;
    hrb[p] = f32_to_bf16(hr);
    hib[p] = f32_to_bf16(hi);
  }
}

// ---------------------------------------------------------------------------
// K6: decode drift = Re(h @ Dec) via WMMA, + residual; write x2 to d_out and
//     emit bf16 MoE tokens tok[token][c] (c<64 real, c>=64 imag)
// ---------------------------------------------------------------------------
__global__ void __launch_bounds__(128) k_decode(
    const u16* __restrict__ hrb, const u16* __restrict__ hib,
    const float* __restrict__ decr, const float* __restrict__ deci,
    const float* __restrict__ x1r, const float* __restrict__ x1i,
    float* __restrict__ out, u16* __restrict__ tokb) {
  __shared__ alignas(16) u16 ldrt[4096], ldnit[4096];   // transposed [n][k]
  int tid = threadIdx.x, wid = tid >> 5, lane = tid & 31;
  for (int i = tid; i < 4096; i += 128) {
    int n = i >> 6, k = i & 63;
    ldrt[i]  = f32_to_bf16(decr[k * D_ + n]);
    ldnit[i] = f32_to_bf16(-deci[k * D_ + n]);
  }
  __syncthreads();
  int rowbase = (blockIdx.x * 4 + wid) * 16;

  v8f acc[4];
#pragma unroll
  for (int i = 0; i < 4; ++i) acc[i] = zero8();

  for (int kc = 0; kc < 2; ++kc) {
    int k0 = kc * 32;
    v16bf ar = fragA_rm(hrb + (size_t)rowbase * D_ + k0, D_, lane);
    v16bf ai = fragA_rm(hib + (size_t)rowbase * D_ + k0, D_, lane);
#pragma unroll
    for (int nb = 0; nb < 4; ++nb) {
      v16bf br  = fragB_tr(ldrt  + nb * 16 * D_ + k0, D_, lane);
      v16bf bni = fragB_tr(ldnit + nb * 16 * D_ + k0, D_, lane);
      acc[nb] = wmma_bf16(ar, br,  acc[nb]);   // hr*Dr
      acc[nb] = wmma_bf16(ai, bni, acc[nb]);   // -hi*Di
    }
  }
  int ncol = lane & 15, mhi = (lane >> 4) * 8;
#pragma unroll
  for (int nb = 0; nb < 4; ++nb) {
    int d = nb * 16 + ncol;
#pragma unroll
    for (int v = 0; v < 8; ++v) {
      int token = rowbase + mhi + v;
      int nseq = token >> 4, t = token & 15;
      int bidx = nseq >> 12, hw = nseq & 4095;
      size_t addr = ((size_t)(bidx * T_ + t) * D_ + d) * HW_ + hw;
      float re = acc[nb][v] + x1r[addr];
      float im = x1i[addr];
      out[addr]            = re;
      out[OUTHALF + addr]  = im;
      tokb[(size_t)token * CH_ + d]      = f32_to_bf16(re);
      tokb[(size_t)token * CH_ + D_ + d] = f32_to_bf16(im);
    }
  }
}

// ---------------------------------------------------------------------------
// K7a: router softmax gates (vectorized token reads)
// ---------------------------------------------------------------------------
__global__ void k_gates(const u16* __restrict__ tokb, const float* __restrict__ rw,
                        const float* __restrict__ rb, float* __restrict__ gates) {
  int token = blockIdx.x * blockDim.x + threadIdx.x;
  float lg[4] = {rb[0], rb[1], rb[2], rb[3]};
  const u16x8* tp = (const u16x8*)(tokb + (size_t)token * CH_);
  for (int c8 = 0; c8 < 16; ++c8) {
    u16x8 vv = tp[c8];
#pragma unroll
    for (int j = 0; j < 8; ++j) {
      float tv = bf16_to_f32(vv[j]);
      int c = c8 * 8 + j;
#pragma unroll
      for (int e = 0; e < 4; ++e) lg[e] += tv * rw[c * 4 + e];
    }
  }
  float mx = fmaxf(fmaxf(lg[0], lg[1]), fmaxf(lg[2], lg[3]));
  float s = 0.f;
#pragma unroll
  for (int e = 0; e < 4; ++e) { lg[e] = expf(lg[e] - mx); s += lg[e]; }
  float invs = 1.f / s;
#pragma unroll
  for (int e = 0; e < 4; ++e) gates[(size_t)token * 4 + e] = lg[e] * invs;
}

// ---------------------------------------------------------------------------
// K7: fused MoE FFN. Gate folded into hdn rows so expert GEMM2 outputs
//     accumulate directly in WMMA accumulators across experts. Full 128x128
//     expert weight staged transposed in LDS once per GEMM (4 barriers/expert).
// ---------------------------------------------------------------------------
__global__ void __launch_bounds__(128) k_moe(
    const u16* __restrict__ tokb, const float* __restrict__ gates,
    const float* __restrict__ w1, const float* __restrict__ b1,
    const float* __restrict__ w2, const float* __restrict__ b2,
    float* __restrict__ out) {
  __shared__ alignas(16) u16 lh[64 * HID_];      // gated gelu(hdn) in bf16
  __shared__ alignas(16) u16 lwt[128 * 128];     // transposed weights [n][k]
  int tid = threadIdx.x, wid = tid >> 5, lane = tid & 31;
  int blkrow = blockIdx.x * 64;
  int rowbase = blkrow + wid * 16;
  int ncol = lane & 15, mhi = (lane >> 4) * 8;

  v8f accO[8];
#pragma unroll
  for (int i = 0; i < 8; ++i) accO[i] = zero8();

  for (int e = 0; e < 4; ++e) {
    v8f acc1[8];
#pragma unroll
    for (int i = 0; i < 8; ++i) acc1[i] = zero8();
    // stage full w1[e] transposed (thread = col n; coalesced reads per k)
    {
      const float* wsrc = w1 + (size_t)e * CH_ * HID_ + tid;
#pragma unroll 8
      for (int k = 0; k < 128; ++k)
        lwt[tid * 128 + k] = f32_to_bf16(wsrc[(size_t)k * HID_]);
    }
    __syncthreads();
#pragma unroll
    for (int kc = 0; kc < 4; ++kc) {
      int k0 = kc * 32;
      v16bf a = fragA_rm(tokb + (size_t)rowbase * CH_ + k0, CH_, lane);
#pragma unroll
      for (int nb = 0; nb < 8; ++nb) {
        v16bf bm = fragB_tr(lwt + nb * 16 * 128 + k0, 128, lane);
        acc1[nb] = wmma_bf16(a, bm, acc1[nb]);
      }
    }
    // epilogue 1: bias + tanh-gelu + gate scale -> LDS (wave-private rows)
#pragma unroll
    for (int nb = 0; nb < 8; ++nb) {
      int hcol = nb * 16 + ncol;
      float bb = b1[e * HID_ + hcol];
#pragma unroll
      for (int v = 0; v < 8; ++v) {
        int lrow = wid * 16 + mhi + v;
        int token = blkrow + lrow;
        float val = acc1[nb][v] + bb;
        float gx = 0.7978845608f * (val + 0.044715f * val * val * val);
        float gl = 0.5f * val * (1.f + tanhf(gx));
        gl *= gates[(size_t)token * 4 + e];
        lh[lrow * HID_ + hcol] = f32_to_bf16(gl);
      }
    }
    __syncthreads();    // all waves done reading w1 fragments
    // stage full w2[e] transposed
    {
      const float* wsrc = w2 + (size_t)e * HID_ * CH_ + tid;
#pragma unroll 8
      for (int k = 0; k < 128; ++k)
        lwt[tid * 128 + k] = f32_to_bf16(wsrc[(size_t)k * CH_]);
    }
    __syncthreads();
#pragma unroll
    for (int kc = 0; kc < 4; ++kc) {
      int k0 = kc * 32;
      v16bf a = fragA_rm(lh + wid * 16 * HID_ + k0, HID_, lane);
#pragma unroll
      for (int nb = 0; nb < 8; ++nb) {
        v16bf bm = fragB_tr(lwt + nb * 16 * 128 + k0, 128, lane);
        accO[nb] = wmma_bf16(a, bm, accO[nb]);
      }
    }
    __syncthreads();    // protect lwt restage next expert
  }
  // final epilogue: + sum_e g_e*b2_e, add delta into d_out (x2 written by K6)
#pragma unroll
  for (int nb = 0; nb < 8; ++nb) {
    int c = nb * 16 + ncol;
    int d = c & 63;
#pragma unroll
    for (int v = 0; v < 8; ++v) {
      int token = rowbase + mhi + v;
      float bsum = 0.f;
#pragma unroll
      for (int e = 0; e < 4; ++e)
        bsum += gates[(size_t)token * 4 + e] * b2[e * CH_ + c];
      float delta = accO[nb][v] + bsum;
      int nseq = token >> 4, t = token & 15;
      int bidx = nseq >> 12, hw = nseq & 4095;
      size_t addr = ((size_t)(bidx * T_ + t) * D_ + d) * HW_ + hw;
      size_t a2 = (c < D_) ? addr : (OUTHALF + addr);
      out[a2] += delta;
    }
  }
}

// ---------------------------------------------------------------------------
// launcher
// ---------------------------------------------------------------------------
extern "C" void kernel_launch(void* const* d_in, const int* in_sizes, int n_in,
                              void* d_out, int out_size, void* d_ws, size_t ws_size,
                              hipStream_t stream) {
  (void)in_sizes; (void)n_in; (void)out_size; (void)ws_size;
  const float* xr    = (const float*)d_in[0];
  const float* xi    = (const float*)d_in[1];
  const float* dt    = (const float*)d_in[2];
  const float* nzr   = (const float*)d_in[3];
  const float* nzi   = (const float*)d_in[4];
  const float* lnsw  = (const float*)d_in[5];
  const float* lnsb  = (const float*)d_in[6];
  const float* convw = (const float*)d_in[7];
  const float* convb = (const float*)d_in[8];
  const float* lntw  = (const float*)d_in[9];
  const float* lntb  = (const float*)d_in[10];
  const float* lamr  = (const float*)d_in[11];
  const float* lami  = (const float*)d_in[12];
  const float* sbr   = (const float*)d_in[13];
  const float* sbi   = (const float*)d_in[14];
  const float* encr  = (const float*)d_in[15];
  const float* enci  = (const float*)d_in[16];
  const float* decr  = (const float*)d_in[17];
  const float* deci  = (const float*)d_in[18];
  const float* rw    = (const float*)d_in[19];
  const float* rb    = (const float*)d_in[20];
  const float* w1    = (const float*)d_in[21];
  const float* b1    = (const float*)d_in[22];
  const float* w2    = (const float*)d_in[23];
  const float* b2    = (const float*)d_in[24];

  char* ws = (char*)d_ws;
  u16*   xc    = (u16*)  (ws + 0);            // 33.5 MB  (later reused as tok)
  float* x1r   = (float*)(ws + 33554432);     // 33.5 MB
  float* x1i   = (float*)(ws + 67108864);     // 33.5 MB
  u16*   xtr   = (u16*)  (ws + 100663296);    // 16.8 MB  (later reused as h_re)
  u16*   xti   = (u16*)  (ws + 117440512);    // 16.8 MB  (later reused as h_im)
  float* ur    = (float*)(ws + 134217728);    // 33.5 MB
  float* ui    = (float*)(ws + 167772160);    // 33.5 MB
  float* gates = (float*)(ws + 201326592);    // 2 MB
  u16* tokb = xc;    // xc dead after conv
  u16* hrb  = xtr;   // xt dead after encode
  u16* hib  = xti;
  float* out = (float*)d_out;

  k_ln_spatial <<<512,  256, 0, stream>>>(xr, xi, lnsw, lnsb, xc);
  k_conv       <<<2048, 128, 0, stream>>>(xc, convw, convb, xr, xi, x1r, x1i);
  k_ln_temporal<<<512,  256, 0, stream>>>(x1r, x1i, lntw, lntb, xtr, xti);
  k_encode     <<<2048, 128, 0, stream>>>(xtr, xti, encr, enci, dt, lamr, lami,
                                          sbr, sbi, nzr, nzi, ur, ui);
  k_scan       <<<2048, 256, 0, stream>>>(ur, ui, dt, lamr, lami, hrb, hib);
  k_decode     <<<2048, 128, 0, stream>>>(hrb, hib, decr, deci, x1r, x1i, out, tokb);
  k_gates      <<<512,  256, 0, stream>>>(tokb, rw, rb, gates);
  k_moe        <<<2048, 128, 0, stream>>>(tokb, gates, w1, b1, w2, b2, out);
}
